// DistributedRealSHT_17549236371791
// MI455X (gfx1250) — matrix operand order repaired and checked by reading, not compiled, not run on hardware
//
#include <hip/hip_runtime.h>
#include <hip/hip_bf16.h>

typedef __attribute__((ext_vector_type(8)))  float          v8f;
typedef __attribute__((ext_vector_type(8)))  __bf16         v8bf;
typedef __attribute__((ext_vector_type(16))) __bf16         v16bf;
typedef __attribute__((ext_vector_type(8)))  unsigned short v8us;

#define NLAT 361
#define NLON 720
#define LMAX 361
#define MMAX 361
#define NCH  256
#define NROWS (NCH * NLAT)       // 92416 rows (c,lat), = 1444*64
#define XK   736                 // NLON padded to 23*32 (zero-filled)
#define LK   384                 // NLAT padded to 12*32 (zero-filled)
#define KSTEP 32
#define LDSP 40                  // LDS row stride in bf16 elems (80B, 16B-aligned)
#define TM   384                 // mode/l row padding (so async copies need no guards)

static constexpr double PI_D = 3.14159265358979323846;

// ---------------------------------------------------------------------------
// fp32 -> bf16 hi/lo Dekker split (RNE on both halves, ~16 mantissa bits kept)
// ---------------------------------------------------------------------------
__device__ __forceinline__ void split_f32(float f, unsigned short& h, unsigned short& l) {
    unsigned u  = __float_as_uint(f);
    unsigned hu = (u + 0x7FFFu + ((u >> 16) & 1u)) & 0xFFFF0000u;
    float    fh = __uint_as_float(hu);
    float    fl = f - fh;                       // exact residual
    unsigned v  = __float_as_uint(fl);
    h = (unsigned short)(hu >> 16);
    l = (unsigned short)((v + 0x7FFFu + ((v >> 16) & 1u)) >> 16);
}

// Assemble a v16bf WMMA fragment from two 16B chunks of an LDS row.
// A-frag (16x32): chunks at (h*8) and (16+h*8).  B-frag (32x16): same (h*8),(16+h*8).
__device__ __forceinline__ v16bf ld_frag(const unsigned short* row, int o0, int o1) {
    v8bf c0 = *(const v8bf*)(row + o0);
    v8bf c1 = *(const v8bf*)(row + o1);
    return __builtin_shufflevector(c0, c1, 0,1,2,3,4,5,6,7,8,9,10,11,12,13,14,15);
}

#define WMMA_BF16(A, B, C) \
    __builtin_amdgcn_wmma_f32_16x16x32_bf16(false, (A), false, (B), (short)0, (C), false, false)

// ---------------------------------------------------------------------------
// Async global->LDS 16B copy (CDNA5 GLOBAL_LOAD_ASYNC_TO_LDS_B128, ASYNCcnt).
// Generic LDS pointers carry the LDS byte offset in their low 32 bits.
// ---------------------------------------------------------------------------
__device__ __forceinline__ void async_cp16(const unsigned short* g, const unsigned short* l) {
    asm volatile("global_load_async_to_lds_b128 %0, %1, off"
                 :: "v"((unsigned)(size_t)l),
                    "v"((unsigned long long)(size_t)g)
                 : "memory");
}
#define WAIT_ASYNC(n) asm volatile("s_wait_asynccnt %0" :: "i"(n) : "memory")

// ---------------------------------------------------------------------------
// Kernel 0: DFT tables, bf16-pair, padded to TM x XK with zeros.
//   Tc = s*cos(-2pi m n/NLON), Ts = s*sin(-2pi m n/NLON), s = 2pi/NLON.
// ---------------------------------------------------------------------------
__global__ void build_dft_tables_split(unsigned short* __restrict__ Tch,
                                       unsigned short* __restrict__ Tcl,
                                       unsigned short* __restrict__ Tsh,
                                       unsigned short* __restrict__ Tsl) {
    int idx = blockIdx.x * blockDim.x + threadIdx.x;
    if (idx >= TM * XK) return;
    int m = idx / XK, n = idx % XK;
    float fc = 0.0f, fs = 0.0f;
    if (m < MMAX && n < NLON) {
        double ang = -2.0 * PI_D * (double)m * (double)n / (double)NLON;
        double s = 2.0 * PI_D / (double)NLON;
        fc = (float)(s * cos(ang));
        fs = (float)(s * sin(ang));
    }
    split_f32(fc, Tch[idx], Tcl[idx]);
    split_f32(fs, Tsh[idx], Tsl[idx]);
}

// ---------------------------------------------------------------------------
// Kernel 0b: x -> bf16-pair with K padding (rows exact, K 720 -> 736).
// ---------------------------------------------------------------------------
__global__ void split_pad_x(const float* __restrict__ src,
                            unsigned short* __restrict__ hi,
                            unsigned short* __restrict__ lo) {
    long n = (long)NROWS * XK;
    for (long idx = (long)blockIdx.x * blockDim.x + threadIdx.x; idx < n;
         idx += (long)gridDim.x * blockDim.x) {
        long row = idx / XK;
        int  k   = (int)(idx - row * XK);
        float f  = (k < NLON) ? src[row * (long)NLON + k] : 0.0f;
        split_f32(f, hi[idx], lo[idx]);
    }
}

// ---------------------------------------------------------------------------
// Kernel 0c: W -> bf16-pair planes [m][TM][LK], zero-padded rows and K.
// ---------------------------------------------------------------------------
__global__ void split_pad_w(const float* __restrict__ src,
                            unsigned short* __restrict__ hi,
                            unsigned short* __restrict__ lo) {
    long n = (long)MMAX * TM * LK;
    for (long idx = (long)blockIdx.x * blockDim.x + threadIdx.x; idx < n;
         idx += (long)gridDim.x * blockDim.x) {
        long m   = idx / ((long)TM * LK);
        long rem = idx - m * (long)TM * LK;
        int  l   = (int)(rem / LK);
        int  k   = (int)(rem - (long)l * LK);
        float f  = (l < LMAX && k < NLAT)
                 ? src[(m * LMAX + l) * (long)NLAT + k] : 0.0f;
        split_f32(f, hi[idx], lo[idx]);
    }
}

// ---------------------------------------------------------------------------
// Kernel 1: DFT as GEMM, bf16-pair, async double-buffered staging.
// M = modes (384 padded), N = 64 rows/block, K = XK.  blockIdx.y = comp.
// 8 waves as 4(M) x 2(N); wave tile 96x32 -> 6x2 f32 accumulators.
// Per-wave async ops per stage: A 6*2 + B 1*2 = 14.
// ---------------------------------------------------------------------------
#define D_BM 384
#define D_BN 64
#define D_OPS 14
#define D_NS (XK / KSTEP)   // 23

__global__ __launch_bounds__(256) void dft_gemm_bf(
        const unsigned short* __restrict__ xh, const unsigned short* __restrict__ xl,
        const unsigned short* __restrict__ Tch, const unsigned short* __restrict__ Tcl,
        const unsigned short* __restrict__ Tsh, const unsigned short* __restrict__ Tsl,
        unsigned short* __restrict__ xfre_h, unsigned short* __restrict__ xfre_l,
        unsigned short* __restrict__ xfim_h, unsigned short* __restrict__ xfim_l) {
    __shared__ unsigned short Ash[2][D_BM * LDSP];
    __shared__ unsigned short Asl[2][D_BM * LDSP];
    __shared__ unsigned short Bsh[2][D_BN * LDSP];
    __shared__ unsigned short Bsl[2][D_BN * LDSP];

    const int comp = blockIdx.y;
    const unsigned short* __restrict__ Ah = comp ? Tsh : Tch;
    const unsigned short* __restrict__ Al = comp ? Tsl : Tcl;
    unsigned short* __restrict__ Oh = comp ? xfim_h : xfre_h;
    unsigned short* __restrict__ Ol = comp ? xfim_l : xfre_l;

    const int n0   = blockIdx.x * D_BN;
    const int t    = threadIdx.x;
    const int lane = t & 31;
    const int wave = t >> 5;
    const int wm   = (wave & 3) * 96;
    const int wn   = (wave >> 2) * 32;
    const int h    = lane >> 4;
    const int ml   = lane & 15;

    v8f acc[6][2] = {};

    auto stage = [&](int buf, int k0) {
        // A tile (384 x 32, hi+lo): unconditional (rows pre-padded).
        for (int i = t; i < D_BM * 4; i += 256) {
            int r = i >> 2, cq = i & 3;
            size_t g = (size_t)r * XK + k0 + cq * 8;
            int    d = r * LDSP + cq * 8;
            async_cp16(Ah + g, &Ash[buf][d]);
            async_cp16(Al + g, &Asl[buf][d]);
        }
        // B tile (64 x 32, hi+lo).
        for (int i = t; i < D_BN * 4; i += 256) {
            int r = i >> 2, cq = i & 3;
            size_t g = (size_t)(n0 + r) * XK + k0 + cq * 8;
            int    d = r * LDSP + cq * 8;
            async_cp16(xh + g, &Bsh[buf][d]);
            async_cp16(xl + g, &Bsl[buf][d]);
        }
    };

    stage(0, 0);
    int buf = 0;
    for (int s = 0; s < D_NS; ++s) {
        if (s + 1 < D_NS) { stage(buf ^ 1, (s + 1) * KSTEP); WAIT_ASYNC(D_OPS); }
        else              { WAIT_ASYNC(0); }
        __syncthreads();

        #pragma unroll
        for (int i = 0; i < 6; ++i) {
            const unsigned short* ra = &Ash[buf][(wm + i * 16 + ml) * LDSP];
            const unsigned short* rl = &Asl[buf][(wm + i * 16 + ml) * LDSP];
            v16bf a_hi = ld_frag(ra, h * 8, 16 + h * 8);
            v16bf a_lo = ld_frag(rl, h * 8, 16 + h * 8);
            #pragma unroll
            for (int j = 0; j < 2; ++j) {
                const unsigned short* rbh = &Bsh[buf][(wn + j * 16 + ml) * LDSP];
                const unsigned short* rbl = &Bsl[buf][(wn + j * 16 + ml) * LDSP];
                v16bf b_hi = ld_frag(rbh, h * 8, 16 + h * 8);
                v16bf b_lo = ld_frag(rbl, h * 8, 16 + h * 8);
                acc[i][j] = WMMA_BF16(a_hi, b_hi, acc[i][j]);
                acc[i][j] = WMMA_BF16(a_hi, b_lo, acc[i][j]);
                acc[i][j] = WMMA_BF16(a_lo, b_hi, acc[i][j]);
            }
        }
        __syncthreads();   // all reads of 'buf' done before it is refilled
        buf ^= 1;
    }

    // Store: D layout N = lane&15, M = r + 8*(lane>>4). Split to bf16 planes
    // laid out [m][c][LK] so the Legendre GEMM reads K-contiguous rows.
    const int nl = lane & 15;
    #pragma unroll
    for (int i = 0; i < 6; ++i) {
        #pragma unroll
        for (int j = 0; j < 2; ++j) {
            int gmBase = wm + i * 16 + 8 * h;
            int gn = n0 + wn + j * 16 + nl;
            int c = gn / NLAT, lat = gn - c * NLAT;
            #pragma unroll
            for (int r = 0; r < 8; ++r) {
                int gm = gmBase + r;
                if (gm < MMAX) {
                    size_t o = (size_t)gm * (NCH * LK) + (size_t)c * LK + lat;
                    split_f32(acc[i][j][r], Oh[o], Ol[o]);
                }
            }
        }
    }
}

// ---------------------------------------------------------------------------
// Kernel 2: per-mode Legendre GEMM, bf16-pair, both components per block
// (shared A = W[m] tile), async double-buffered staging.
// M = LMAX (tiles of 128, rows pre-padded to 384), N = NCH (tiles of 64),
// K = LK.  grid = (3*4, MMAX).  Per-wave async ops per stage: 4 + 4 = 8.
// ---------------------------------------------------------------------------
#define L_BM 128
#define L_BN 64
#define L_OPS 8
#define L_NS (LK / KSTEP)   // 12

__global__ __launch_bounds__(256) void leg_gemm_bf(
        const unsigned short* __restrict__ Wh, const unsigned short* __restrict__ Wl,
        const unsigned short* __restrict__ xfre_h, const unsigned short* __restrict__ xfre_l,
        const unsigned short* __restrict__ xfim_h, const unsigned short* __restrict__ xfim_l,
        float* __restrict__ out) {
    __shared__ unsigned short Ash[2][L_BM * LDSP];
    __shared__ unsigned short Asl[2][L_BM * LDSP];
    __shared__ unsigned short Bsh[2][2][L_BN * LDSP];
    __shared__ unsigned short Bsl[2][2][L_BN * LDSP];

    const int m  = blockIdx.y;
    const int l0 = (blockIdx.x % 3) * L_BM;
    const int n0 = (blockIdx.x / 3) * L_BN;

    const unsigned short* __restrict__ Awh = Wh + (size_t)m * TM * LK;
    const unsigned short* __restrict__ Awl = Wl + (size_t)m * TM * LK;
    const unsigned short* __restrict__ Bph[2] = { xfre_h + (size_t)m * (NCH * LK),
                                                  xfim_h + (size_t)m * (NCH * LK) };
    const unsigned short* __restrict__ Bpl[2] = { xfre_l + (size_t)m * (NCH * LK),
                                                  xfim_l + (size_t)m * (NCH * LK) };

    const int t    = threadIdx.x;
    const int lane = t & 31;
    const int wave = t >> 5;
    const int wm   = (wave & 3) * 32;
    const int wn   = (wave >> 2) * 32;
    const int h    = lane >> 4;
    const int ml   = lane & 15;

    v8f acc[2][2][2] = {};   // [i][j][comp]

    auto stage = [&](int buf, int k0) {
        for (int i = t; i < L_BM * 4; i += 256) {
            int r = i >> 2, cq = i & 3;
            size_t g = (size_t)(l0 + r) * LK + k0 + cq * 8;  // rows pre-padded
            int    d = r * LDSP + cq * 8;
            async_cp16(Awh + g, &Ash[buf][d]);
            async_cp16(Awl + g, &Asl[buf][d]);
        }
        for (int i = t; i < L_BN * 4 * 2; i += 256) {
            int cc = i & 1, r = (i >> 1) >> 2, cq = (i >> 1) & 3;
            size_t g = (size_t)(n0 + r) * LK + k0 + cq * 8;
            int    d = r * LDSP + cq * 8;
            async_cp16(Bph[cc] + g, &Bsh[buf][cc][d]);
            async_cp16(Bpl[cc] + g, &Bsl[buf][cc][d]);
        }
    };

    stage(0, 0);
    int buf = 0;
    for (int s = 0; s < L_NS; ++s) {
        if (s + 1 < L_NS) { stage(buf ^ 1, (s + 1) * KSTEP); WAIT_ASYNC(L_OPS); }
        else              { WAIT_ASYNC(0); }
        __syncthreads();

        #pragma unroll
        for (int i = 0; i < 2; ++i) {
            const unsigned short* ra = &Ash[buf][(wm + i * 16 + ml) * LDSP];
            const unsigned short* rl = &Asl[buf][(wm + i * 16 + ml) * LDSP];
            v16bf a_hi = ld_frag(ra, h * 8, 16 + h * 8);
            v16bf a_lo = ld_frag(rl, h * 8, 16 + h * 8);
            #pragma unroll
            for (int j = 0; j < 2; ++j) {
                #pragma unroll
                for (int cc = 0; cc < 2; ++cc) {
                    const unsigned short* rbh = &Bsh[buf][cc][(wn + j * 16 + ml) * LDSP];
                    const unsigned short* rbl = &Bsl[buf][cc][(wn + j * 16 + ml) * LDSP];
                    v16bf b_hi = ld_frag(rbh, h * 8, 16 + h * 8);
                    v16bf b_lo = ld_frag(rbl, h * 8, 16 + h * 8);
                    acc[i][j][cc] = WMMA_BF16(a_hi, b_hi, acc[i][j][cc]);
                    acc[i][j][cc] = WMMA_BF16(a_hi, b_lo, acc[i][j][cc]);
                    acc[i][j][cc] = WMMA_BF16(a_lo, b_hi, acc[i][j][cc]);
                }
            }
        }
        __syncthreads();
        buf ^= 1;
    }

    // out[((c*LMAX + l)*MMAX + m)*2 + comp], complex64 interleaved.
    const int nl = lane & 15;
    #pragma unroll
    for (int i = 0; i < 2; ++i) {
        #pragma unroll
        for (int j = 0; j < 2; ++j) {
            int glBase = l0 + wm + i * 16 + 8 * h;
            int gc = n0 + wn + j * 16 + nl;
            #pragma unroll
            for (int r = 0; r < 8; ++r) {
                int gl = glBase + r;
                if (gl < LMAX) {
                    size_t o = (((size_t)gc * LMAX + gl) * MMAX + m) * 2;
                    out[o + 0] = acc[i][j][0][r];
                    out[o + 1] = acc[i][j][1][r];
                }
            }
        }
    }
}

// ---------------------------------------------------------------------------
extern "C" void kernel_launch(void* const* d_in, const int* in_sizes, int n_in,
                              void* d_out, int out_size, void* d_ws, size_t ws_size,
                              hipStream_t stream) {
    const float* x = (const float*)d_in[0];   // (1,256,361,720) fp32
    const float* w = (const float*)d_in[1];   // (361,361,361)   fp32
    float* out = (float*)d_out;               // (1,256,361,361) c64 interleaved

    // Workspace layout (ushort = bf16 bits):
    unsigned short* p = (unsigned short*)d_ws;
    unsigned short* Tch = p; p += (size_t)TM * XK;
    unsigned short* Tcl = p; p += (size_t)TM * XK;
    unsigned short* Tsh = p; p += (size_t)TM * XK;
    unsigned short* Tsl = p; p += (size_t)TM * XK;
    unsigned short* xh  = p; p += (size_t)NROWS * XK;
    unsigned short* xl  = p; p += (size_t)NROWS * XK;
    unsigned short* Wh  = p; p += (size_t)MMAX * TM * LK;
    unsigned short* Wl  = p; p += (size_t)MMAX * TM * LK;
    unsigned short* xfre_h = p; p += (size_t)MMAX * NCH * LK;
    unsigned short* xfre_l = p; p += (size_t)MMAX * NCH * LK;
    unsigned short* xfim_h = p; p += (size_t)MMAX * NCH * LK;
    unsigned short* xfim_l = p; p += (size_t)MMAX * NCH * LK;

    // 0) DFT tables (bf16-pair, row+K padded)
    {
        int n = TM * XK;
        build_dft_tables_split<<<(n + 255) / 256, 256, 0, stream>>>(Tch, Tcl, Tsh, Tsl);
    }
    // 0b) Pre-split x and W into bf16-pair planes with padded strides.
    {
        long n = (long)NROWS * XK;
        split_pad_x<<<(int)((n + 255) / 256), 256, 0, stream>>>(x, xh, xl);
    }
    {
        long n = (long)MMAX * TM * LK;
        split_pad_w<<<(int)((n + 255) / 256), 256, 0, stream>>>(w, Wh, Wl);
    }
    // 1) DFT GEMM: all modes per block; grid.x over rows, grid.y = comp.
    {
        dim3 grid(NROWS / D_BN, 2);
        dft_gemm_bf<<<grid, 256, 0, stream>>>(xh, xl, Tch, Tcl, Tsh, Tsl,
                                              xfre_h, xfre_l, xfim_h, xfim_l);
    }
    // 2) Legendre GEMMs: both components per block, one m per grid.y slice.
    {
        dim3 grid(3 * (NCH / L_BN), MMAX);
        leg_gemm_bf<<<grid, 256, 0, stream>>>(Wh, Wl, xfre_h, xfre_l,
                                              xfim_h, xfim_l, out);
    }
}
